// ChannelSqueezeSpatialAttention_58025008169445
// MI455X (gfx1250) — matrix operand order repaired and checked by reading, not compiled, not run on hardware
//
#include <hip/hip_runtime.h>
#include <hip/hip_bf16.h>

// ---------------------------------------------------------------------------
// ChannelSqueezeSpatialAttention on gfx1250 (MI455X), wave32 WMMA bf16 path.
// Pipeline: 3x projection GEMM -> 2x LDS-staged squeeze-conv (implicit GEMM)
//           -> fused flash-style attention (online softmax, no sim tensor).
// All matrix math through v_wmma_f32_16x16x32_bf16.
// ---------------------------------------------------------------------------

typedef __attribute__((ext_vector_type(16))) __bf16 v16bf;
typedef __attribute__((ext_vector_type(8)))  float  v8f;

#define WMMA_BF16(A, B, C) \
    __builtin_amdgcn_wmma_f32_16x16x32_bf16(false, (A), false, (B), (short)0, (C), false, false)

__device__ __forceinline__ unsigned short f2bf_bits(float f) {
    union { __bf16 b; unsigned short u; } cv;
    cv.b = (__bf16)f;
    return cv.u;
}

// ---------------------------------------------------------------------------
// Projection GEMM:  Y[b][m][n] = sum_k W[m][k] * X[b][k][n]
// M = K = 256 (channels), N = spatial (4096 or 1024). One wave per 16x16 tile.
// ---------------------------------------------------------------------------
__global__ __launch_bounds__(32)
void csa_proj_gemm(const float* __restrict__ W, const float* __restrict__ X,
                   float* __restrict__ Y, int N) {
    const int lane = threadIdx.x;
    const int lo = lane & 15, hi = lane >> 4;
    const int nt = blockIdx.x, mt = blockIdx.y, b = blockIdx.z;

    const float* Xb = X + (size_t)b * 256 * N;
    float*       Yb = Y + (size_t)b * 256 * N;
    const int m = mt * 16 + lo;   // A row for this lane
    const int n = nt * 16 + lo;   // B/C column for this lane

    v8f acc = {};
    for (int k0 = 0; k0 < 256; k0 += 32) {
        // A tile (16x32 bf16): lane holds row m, K = hi*16 + e
        v16bf a;
        const float* arow = W + (size_t)m * 256 + k0 + hi * 16;
#pragma unroll
        for (int e = 0; e < 8; ++e) {
            a[e]     = (__bf16)arow[e];
            a[e + 8] = (__bf16)arow[8 + e];
        }
        // B tile (32x16 bf16): lane holds col n, K = hi*16 + i
        v16bf bm;
#pragma unroll
        for (int i = 0; i < 16; ++i)
            bm[i] = (__bf16)Xb[(size_t)(k0 + hi * 16 + i) * N + n];
        acc = WMMA_BF16(a, bm, acc);
    }
    // C layout: VGPR j -> row (j + 8*hi), col n
#pragma unroll
    for (int j = 0; j < 8; ++j)
        Yb[(size_t)(mt * 16 + j + 8 * hi) * N + n] = acc[j];
}

// ---------------------------------------------------------------------------
// Squeeze conv as 147 accumulated implicit GEMMs, LDS-staged operands.
// O[b][nh][cg][y*W+x] = sum_{d,kd,ky,kx} P[b][(nh+kd-1)*64+d][y+ky-3][x+kx-3]
//                                        * Wt[cg][d][kd][ky][kx]
// Block: 128 threads = 4 waves = {cg tile 0/1} x {x subtile 0/1}, one 32-wide
// output strip per block. Input patch staged per kd, weight slice per (kd,ky).
// ---------------------------------------------------------------------------
#define IB_DP 72                       // padded d-stride (bank spread, 16B align)
#define IB_ELEMS (7 * 38 * IB_DP)      // 19152 bf16 (~37.4 KB)
#define WB_ELEMS (7 * 32 * IB_DP)      // 16128 bf16 (~31.5 KB)

__global__ __launch_bounds__(128)
void csa_squeeze_conv(const float* __restrict__ P, const float* __restrict__ Wt,
                      float* __restrict__ O, int H, int Wd) {
    __shared__ __attribute__((aligned(16))) __bf16 ibuf[IB_ELEMS]; // [ry][cx][d]
    __shared__ __attribute__((aligned(16))) __bf16 wbuf[WB_ELEMS]; // [kx][cg][d]

    const int tid  = threadIdx.x;
    const int lane = tid & 31;
    const int lo = lane & 15, hi = lane >> 4;
    const int wv   = tid >> 5;
    const int mt   = wv >> 1;          // cg tile (0..1)
    const int xs   = wv & 1;           // x subtile (0..1)
    const int bz   = blockIdx.z;       // b*4 + nh
    const int b = bz >> 2, nh = bz & 3;
    const int tilesPerRow = Wd >> 5;   // 32-wide strips per row
    const int y  = blockIdx.x / tilesPerRow;
    const int x0 = (blockIdx.x % tilesPerRow) * 32;

    const int cgl = mt * 16 + lo;      // A row (17 valid of 32)
    const float* Pb = P + (size_t)b * 256 * H * Wd;

    v8f acc = {};
    for (int kd = 0; kd < 3; ++kd) {
        const int nh_in = nh + kd - 1;
        if (nh_in < 0 || nh_in >= 4) continue;   // uniform across block
        const float* Pc = Pb + (size_t)nh_in * 64 * H * Wd;

        // Prefetch next kd's slice into L2 while we work on this one.
        if (nh_in + 1 < 4 && kd < 2)
            __builtin_prefetch(Pc + (size_t)64 * H * Wd + (size_t)tid * 64, 0, 1);

        __syncthreads();               // previous-iteration readers done
        // ---- stage input patch: 64ch x 7row x 38col -> bf16 LDS ----
        for (int s = tid; s < 64 * 7 * 38; s += 128) {
            const int d  = s / 266;    // 266 = 7*38
            const int r  = s % 266;
            const int ry = r / 38;
            const int cx = r % 38;
            const int yy = y + ry - 3;
            const int xx = x0 + cx - 3;
            const bool ok = (yy >= 0) && (yy < H) && (xx >= 0) && (xx < Wd);
            ibuf[(ry * 38 + cx) * IB_DP + d] =
                ok ? (__bf16)Pc[((size_t)d * H + yy) * Wd + xx] : (__bf16)0.0f;
        }

        for (int ky = 0; ky < 7; ++ky) {
            __syncthreads();           // prev weight-slice readers done; ibuf visible
            // ---- stage weight slice for (kd,ky): 32cg x 64d x 7kx ----
            for (int s = tid; s < 32 * 64 * 7; s += 128) {
                const int cg = s / 448;          // 448 = 64*7
                const int rr = s % 448;
                const int d  = rr / 7;
                const int kx = rr % 7;
                wbuf[(kx * 32 + cg) * IB_DP + d] =
                    (cg < 17)
                        ? (__bf16)Wt[(((size_t)cg * 64 + d) * 3 + kd) * 49 + ky * 7 + kx]
                        : (__bf16)0.0f;
            }
            __syncthreads();           // staging visible

            // ---- 14 WMMAs per (kd,ky), all operands from LDS ----
#pragma unroll
            for (int kx = 0; kx < 7; ++kx) {
                const int cxb = (ky * 38 + xs * 16 + lo + kx) * IB_DP;
#pragma unroll
                for (int ds = 0; ds < 2; ++ds) {
                    v16bf a, bm;
                    const int wb = (kx * 32 + cgl) * IB_DP + ds * 32 + hi * 16;
                    __builtin_memcpy(&a, &wbuf[wb], 16);              // K grp 0
                    __builtin_memcpy((char*)&a + 16, &wbuf[wb + 8], 16); // K grp 1
                    __builtin_memcpy(&bm, &ibuf[cxb + ds * 32 + hi * 16], 32);
                    acc = WMMA_BF16(a, bm, acc);
                }
            }
        }
    }

    float* Ob = O + ((size_t)bz * 17) * (size_t)(H * Wd);
#pragma unroll
    for (int j = 0; j < 8; ++j) {
        const int row = mt * 16 + j + 8 * hi;
        if (row < 17)
            Ob[(size_t)row * (H * Wd) + (size_t)y * Wd + x0 + xs * 16 + lo] = acc[j];
    }
}

// ---------------------------------------------------------------------------
// Fused attention, flash-style. One wave handles 16 queries over 1024 keys.
// QC: [b][nh][17][4096]  (cg-major, read transposed as A operand, cg padded->32)
// KC: [b][nh][17][1024]
// VP: [b][256][1024]     (projection layout; head dim d = channel % 64)
// OUT:[b][256][4096]
// ---------------------------------------------------------------------------
__global__ __launch_bounds__(32)
void csa_attention(const float* __restrict__ QC, const float* __restrict__ KC,
                   const float* __restrict__ VP, float* __restrict__ OUT) {
    __shared__ unsigned short pbuf[16 * 32];   // P tile staging (bf16 bits)
    const int lane = threadIdx.x;
    const int lo = lane & 15, hi = lane >> 4;
    const int qt = blockIdx.x;                 // 256 query tiles of 16
    const int bz = blockIdx.y;                 // b*4 + nh
    const int b = bz >> 2, nh = bz & 3;
    const float SCALE = 0.125f;                // 64^-0.5

    const float* qc = QC + (size_t)bz * 17 * 4096;
    const float* kc = KC + (size_t)bz * 17 * 1024;
    const float* vp = VP + ((size_t)b * 256 + nh * 64) * 1024;

    // A operand for scores: 16 queries x 32 (cg padded from 17). Loaded once.
    v16bf aq;
    {
        const int q = qt * 16 + lo;
#pragma unroll
        for (int e = 0; e < 8; ++e) {
            const int c0 = hi * 16 + e;
            const int c1 = c0 + 8;
            aq[e]     = (c0 < 17) ? (__bf16)qc[(size_t)c0 * 4096 + q] : (__bf16)0.0f;
            aq[e + 8] = (c1 < 17) ? (__bf16)qc[(size_t)c1 * 4096 + q] : (__bf16)0.0f;
        }
    }

    v8f acc0 = {}, acc1 = {}, acc2 = {}, acc3 = {};
    float mrow[8], lrow[8];
#pragma unroll
    for (int j = 0; j < 8; ++j) { mrow[j] = -3.0e38f; lrow[j] = 0.0f; }

    for (int ks = 0; ks < 1024; ks += 32) {
        // ---- S = (q.cg) @ (cg.k) for 32 keys: two 16x16 WMMA tiles ----
        v16bf bk0, bk1;
#pragma unroll
        for (int i = 0; i < 16; ++i) {
            const int c = hi * 16 + i;
            const bool ok = (c < 17);
            bk0[i] = ok ? (__bf16)kc[(size_t)c * 1024 + ks + lo]      : (__bf16)0.0f;
            bk1[i] = ok ? (__bf16)kc[(size_t)c * 1024 + ks + 16 + lo] : (__bf16)0.0f;
        }
        v8f z0 = {}, z1 = {};
        v8f s0 = WMMA_BF16(aq, bk0, z0);
        v8f s1 = WMMA_BF16(aq, bk1, z1);

        // ---- online softmax over the 32 new keys ----
        float alpha[8];
#pragma unroll
        for (int j = 0; j < 8; ++j) {
            float a0 = s0[j] * SCALE, a1 = s1[j] * SCALE;
            float t = fmaxf(a0, a1);
            t = fmaxf(t, __shfl_xor(t, 1, 32));
            t = fmaxf(t, __shfl_xor(t, 2, 32));
            t = fmaxf(t, __shfl_xor(t, 4, 32));
            t = fmaxf(t, __shfl_xor(t, 8, 32));      // row max within 16-lane half
            const float mnew = fmaxf(mrow[j], t);
            alpha[j] = __expf(mrow[j] - mnew);
            a0 = __expf(a0 - mnew);
            a1 = __expf(a1 - mnew);
            float r = a0 + a1;
            r += __shfl_xor(r, 1, 32);
            r += __shfl_xor(r, 2, 32);
            r += __shfl_xor(r, 4, 32);
            r += __shfl_xor(r, 8, 32);               // row sum
            lrow[j] = lrow[j] * alpha[j] + r;
            mrow[j] = mnew;
            s0[j] = a0; s1[j] = a1;                  // P values
            acc0[j] *= alpha[j]; acc1[j] *= alpha[j];
            acc2[j] *= alpha[j]; acc3[j] *= alpha[j];
        }

        // ---- C-layout P -> LDS -> A-layout bf16 operand (16q x 32k) ----
#pragma unroll
        for (int j = 0; j < 8; ++j) {
            const int row = j + 8 * hi;
            pbuf[row * 32 + lo]      = f2bf_bits(s0[j]);
            pbuf[row * 32 + 16 + lo] = f2bf_bits(s1[j]);
        }
        asm volatile("s_wait_dscnt 0" ::: "memory");
        union { v16bf v; unsigned short h[16]; } pu;
#pragma unroll
        for (int u = 0; u < 8; ++u) {
            const int K = hi * 16 + u * 2;
            pu.h[2 * u]     = pbuf[lo * 32 + K];
            pu.h[2 * u + 1] = pbuf[lo * 32 + K + 1];
        }
        const v16bf pa = pu.v;
        asm volatile("" ::: "memory");   // keep next iter's LDS stores after reads

        // ---- acc += P(16x32) @ V(32x16) for the 4 head-dim tiles ----
#pragma unroll
        for (int t = 0; t < 4; ++t) {
            v16bf bv;
            const int d = t * 16 + lo;
#pragma unroll
            for (int i = 0; i < 16; ++i)
                bv[i] = (__bf16)vp[(size_t)d * 1024 + ks + hi * 16 + i];
            switch (t) {
                case 0: acc0 = WMMA_BF16(pa, bv, acc0); break;
                case 1: acc1 = WMMA_BF16(pa, bv, acc1); break;
                case 2: acc2 = WMMA_BF16(pa, bv, acc2); break;
                case 3: acc3 = WMMA_BF16(pa, bv, acc3); break;
            }
        }
    }

    // ---- epilogue: OUT[b][nh*64+d][q] = acc / l ----
    float* ob = OUT + ((size_t)b * 256 + nh * 64) * 4096;
#pragma unroll
    for (int j = 0; j < 8; ++j) {
        const int q = qt * 16 + j + 8 * hi;
        const float inv = 1.0f / lrow[j];
        ob[(size_t)(0 * 16 + lo) * 4096 + q]  = acc0[j] * inv;
        ob[(size_t)(1 * 16 + lo) * 4096 + q]  = acc1[j] * inv;
        ob[(size_t)(2 * 16 + lo) * 4096 + q]  = acc2[j] * inv;
        ob[(size_t)(3 * 16 + lo) * 4096 + q]  = acc3[j] * inv;
    }
}

// ---------------------------------------------------------------------------
extern "C" void kernel_launch(void* const* d_in, const int* in_sizes, int n_in,
                              void* d_out, int out_size, void* d_ws, size_t ws_size,
                              hipStream_t stream) {
    (void)in_sizes; (void)n_in; (void)out_size; (void)ws_size;
    const float* xq   = (const float*)d_in[0];  // [4][256][64][64]
    const float* xkv  = (const float*)d_in[1];  // [4][256][32][32]
    const float* wq   = (const float*)d_in[2];  // [256][256]
    const float* wk   = (const float*)d_in[3];
    const float* wv   = (const float*)d_in[4];
    const float* embq = (const float*)d_in[5];  // [17][64][3][7][7]
    const float* embk = (const float*)d_in[6];
    float* out = (float*)d_out;                 // [4][256][4096]

    // Workspace layout (floats): ~30.7 MB total
    float* qproj = (float*)d_ws;                        // 4*256*4096
    float* kproj = qproj + (size_t)4 * 256 * 4096;      // 4*256*1024
    float* vproj = kproj + (size_t)4 * 256 * 1024;      // 4*256*1024
    float* qc    = vproj + (size_t)4 * 256 * 1024;      // 4*4*17*4096
    float* kc    = qc    + (size_t)4 * 4 * 17 * 4096;   // 4*4*17*1024

    csa_proj_gemm<<<dim3(4096 / 16, 16, 4), dim3(32), 0, stream>>>(wq, xq, qproj, 4096);
    csa_proj_gemm<<<dim3(1024 / 16, 16, 4), dim3(32), 0, stream>>>(wk, xkv, kproj, 1024);
    csa_proj_gemm<<<dim3(1024 / 16, 16, 4), dim3(32), 0, stream>>>(wv, xkv, vproj, 1024);
    csa_squeeze_conv<<<dim3(4096 / 32, 1, 16), dim3(128), 0, stream>>>(qproj, embq, qc, 64, 64);
    csa_squeeze_conv<<<dim3(1024 / 32, 1, 16), dim3(128), 0, stream>>>(kproj, embk, kc, 32, 32);
    csa_attention<<<dim3(256, 16), dim3(32), 0, stream>>>(qc, kc, vproj, out);
}
